// Attention_11605001634315
// MI455X (gfx1250) — compile-verified, matrix operand order checked by reading, not run on hardware
//
#include <hip/hip_runtime.h>
#include <hip/hip_bf16.h>
#include <stdint.h>

// Problem constants (B, DIM, N, H from the reference)
#define BB   2
#define DIM  1024
#define NN   2048
#define HH   16
#define HD   64
#define SCALE 0.125f            // HD^-0.5 = 64^-0.5

typedef __bf16 bf16x8   __attribute__((ext_vector_type(8)));
typedef __bf16 bf16x16  __attribute__((ext_vector_type(16)));
typedef float  f32x8    __attribute__((ext_vector_type(8)));
typedef float  f32x4    __attribute__((ext_vector_type(4)));   // for nontemporal st

// ---------------------------------------------------------------------------
// WMMA helpers (CDNA5 wave32, v_wmma_f32_16x16x32_bf16)
// A-fragment layout (16x32 bf16, ISA 7.12.2): lane L (L<16) holds row M=L,
//   VGPR0..3 = K 0..7, VGPR4..7 = K 16..23; lane L+16 holds K 8..15 / 24..31.
// B-fragment = B^T rows in the same pattern (lane%16 = N column).
// ---------------------------------------------------------------------------
__device__ __forceinline__ bf16x16 load_frag_bf16(const __bf16* base, int ld, int lane) {
    const int r = lane & 15, half = lane >> 4;
    const __bf16* p = base + (size_t)r * ld + half * 8;
    bf16x8 lo = *(const bf16x8*)(p);        // K = half*8 + 0..7
    bf16x8 hi = *(const bf16x8*)(p + 16);   // K = 16 + half*8 + 0..7
    return __builtin_shufflevector(lo, hi, 0,1,2,3,4,5,6,7,8,9,10,11,12,13,14,15);
}

__device__ __forceinline__ bf16x16 load_frag_f32(const float* base, int ld, int lane) {
    const int r = lane & 15, half = lane >> 4;
    const float* p = base + (size_t)r * ld + half * 8;
    f32x4 f0 = *(const f32x4*)(p);
    f32x4 f1 = *(const f32x4*)(p + 4);
    f32x4 f2 = *(const f32x4*)(p + 16);
    f32x4 f3 = *(const f32x4*)(p + 20);
    bf16x16 v;
    v[0]=(__bf16)f0.x;  v[1]=(__bf16)f0.y;  v[2]=(__bf16)f0.z;  v[3]=(__bf16)f0.w;
    v[4]=(__bf16)f1.x;  v[5]=(__bf16)f1.y;  v[6]=(__bf16)f1.z;  v[7]=(__bf16)f1.w;
    v[8]=(__bf16)f2.x;  v[9]=(__bf16)f2.y;  v[10]=(__bf16)f2.z; v[11]=(__bf16)f2.w;
    v[12]=(__bf16)f3.x; v[13]=(__bf16)f3.y; v[14]=(__bf16)f3.z; v[15]=(__bf16)f3.w;
    return v;
}

__device__ __forceinline__ f32x8 wmma_bf16(bf16x16 a, bf16x16 b, f32x8 c) {
    return __builtin_amdgcn_wmma_f32_16x16x32_bf16(false, a, false, b, (short)0, c,
                                                   false, false);
}

__device__ __forceinline__ unsigned pack2(float a, float b) {
    unsigned short ua = __builtin_bit_cast(unsigned short, (__bf16)a);
    unsigned short ub = __builtin_bit_cast(unsigned short, (__bf16)b);
    return (unsigned)ua | ((unsigned)ub << 16);
}

// ---------------------------------------------------------------------------
// K1: x (B, DIM, N) f32  ->  xt (B, N, DIM) bf16   (LDS-tiled transpose)
// ---------------------------------------------------------------------------
__global__ void k_transpose_x(const float* __restrict__ x, __bf16* __restrict__ xt) {
    __shared__ float tile[32][33];
    const int b  = blockIdx.z;
    const int n0 = blockIdx.x * 32;
    const int d0 = blockIdx.y * 32;
    const int tx = threadIdx.x, ty = threadIdx.y;   // block (32, 8)
    const float* xb = x + (size_t)b * DIM * NN;
    #pragma unroll
    for (int i = 0; i < 32; i += 8)
        tile[ty + i][tx] = xb[(size_t)(d0 + ty + i) * NN + n0 + tx];
    __syncthreads();
    __bf16* xtb = xt + (size_t)b * NN * DIM;
    #pragma unroll
    for (int i = 0; i < 32; i += 8)
        xtb[(size_t)(n0 + ty + i) * DIM + d0 + tx] = (__bf16)tile[tx][ty + i];
}

// ---------------------------------------------------------------------------
// K2: w_qkv f32 -> bf16
// ---------------------------------------------------------------------------
__global__ void k_convert_w(const float* __restrict__ w, __bf16* __restrict__ wb, int n) {
    int i = blockIdx.x * blockDim.x + threadIdx.x;
    if (i < n) wb[i] = (__bf16)w[i];
}

// ---------------------------------------------------------------------------
// K3: QKV GEMM.  D[j][i] = sum_d w[j][d] * xt[i][d]  (A = w rows, B^T = xt rows)
// Wave macro-tile 32(j) x 64(i): 8 accumulators, 8 WMMAs per 6 fragment loads.
// Straight-line loop body (no branches, no runtime-indexed register arrays):
// the compiler clauses the 12 b128 loads and gates the 8 WMMAs with partial
// s_wait_loadcnt, which overlaps most VMEM latency; 8 waves/block hide the rest.
// ---------------------------------------------------------------------------
__global__ __launch_bounds__(256)
void k_qkv_gemm(const __bf16* __restrict__ xt, const __bf16* __restrict__ wb,
                __bf16* __restrict__ qkv) {
    const int b    = blockIdx.z;
    const int j0   = blockIdx.x * 64;    // output-channel tile of 3*DIM
    const int i0   = blockIdx.y * 256;   // token tile
    const int tid  = threadIdx.x, wave = tid >> 5, lane = tid & 31;
    const int wj   = wave >> 2;          // 0..1 : 32-wide j half
    const int wi   = wave & 3;           // 0..3 : 64-wide i group
    const __bf16* Arow = wb + (size_t)(j0 + wj * 32) * DIM;
    const __bf16* Brow = xt + (size_t)b * NN * DIM + (size_t)(i0 + wi * 64) * DIM;

    f32x8 acc[2][4] = {};
    for (int k0 = 0; k0 < DIM; k0 += 32) {
        bf16x16 a0 = load_frag_bf16(Arow + k0, DIM, lane);
        bf16x16 a1 = load_frag_bf16(Arow + (size_t)16 * DIM + k0, DIM, lane);
        bf16x16 bfr[4];
        #pragma unroll
        for (int t = 0; t < 4; ++t)
            bfr[t] = load_frag_bf16(Brow + (size_t)t * 16 * DIM + k0, DIM, lane);
        #pragma unroll
        for (int t = 0; t < 4; ++t) acc[0][t] = wmma_bf16(a0, bfr[t], acc[0][t]);
        #pragma unroll
        for (int t = 0; t < 4; ++t) acc[1][t] = wmma_bf16(a1, bfr[t], acc[1][t]);
    }

    const int half = lane >> 4, l15 = lane & 15;
    __bf16* qb = qkv + (size_t)b * NN * 3 * DIM;
    #pragma unroll
    for (int m = 0; m < 2; ++m) {
        const int jcol = j0 + wj * 32 + m * 16 + half * 8;   // 8 consecutive j
        #pragma unroll
        for (int t = 0; t < 4; ++t) {
            const int i = i0 + wi * 64 + t * 16 + l15;
            uint4 pk;
            pk.x = pack2(acc[m][t][0], acc[m][t][1]);
            pk.y = pack2(acc[m][t][2], acc[m][t][3]);
            pk.z = pack2(acc[m][t][4], acc[m][t][5]);
            pk.w = pack2(acc[m][t][6], acc[m][t][7]);
            *(uint4*)(qb + (size_t)i * (3 * DIM) + jcol) = pk;
        }
    }
}

// ---------------------------------------------------------------------------
// K4: logits.  D[key][query] = k_z . q_z  -> attn[z][query][key] = D * SCALE
// Wave tile 32 keys x 64 queries; K = HD = 64 -> 2 k-steps, one 6-fragment
// group live at a time.  attn is a 537 MB stream that cannot be L2-resident
// -> non-temporal stores so qkv (25 MB) keeps its L2 residency for K6.
// ---------------------------------------------------------------------------
__global__ __launch_bounds__(256)
void k_logits(const __bf16* __restrict__ qkv, float* __restrict__ attn) {
    const int z  = blockIdx.z;               // z = h*B + b
    const int b  = z % BB, h = z / BB;
    const int m0 = blockIdx.x * 64;          // key tile (block)
    const int q0 = blockIdx.y * 256;         // query tile (block)
    const int tid = threadIdx.x, wave = tid >> 5, lane = tid & 31;
    const int wk = wave >> 2;                // 0..1 : 32-wide key half
    const int wq = wave & 3;                 // 0..3 : 64-wide query group
    // (3n, dim) view of qkv[b]; head slice = 64 contiguous channels
    const __bf16* base = qkv + (size_t)b * NN * 3 * DIM + (size_t)h * HD;
    const __bf16* qm = base;                        // q rows, ld = DIM
    const __bf16* km = base + (size_t)NN * DIM;     // k rows, ld = DIM
    const __bf16* A0 = km + (size_t)(m0 + wk * 32) * DIM;
    const __bf16* B0 = qm + (size_t)(q0 + wq * 64) * DIM;

    f32x8 acc[2][4] = {};
    #pragma unroll
    for (int ks = 0; ks < 2; ++ks) {
        bf16x16 ak[2], qf[4];
        #pragma unroll
        for (int m = 0; m < 2; ++m)
            ak[m] = load_frag_bf16(A0 + (size_t)m * 16 * DIM + ks * 32, DIM, lane);
        #pragma unroll
        for (int t = 0; t < 4; ++t)
            qf[t] = load_frag_bf16(B0 + (size_t)t * 16 * DIM + ks * 32, DIM, lane);
        #pragma unroll
        for (int m = 0; m < 2; ++m)
            #pragma unroll
            for (int t = 0; t < 4; ++t)
                acc[m][t] = wmma_bf16(ak[m], qf[t], acc[m][t]);
    }

    const int half = lane >> 4, l15 = lane & 15;
    float* az = attn + (size_t)z * NN * NN;
    #pragma unroll
    for (int m = 0; m < 2; ++m) {
        const int kcol = m0 + wk * 32 + m * 16 + half * 8;
        #pragma unroll
        for (int t = 0; t < 4; ++t) {
            const int q = q0 + wq * 64 + t * 16 + l15;
            f32x4 s0, s1;
            s0.x = acc[m][t][0] * SCALE; s0.y = acc[m][t][1] * SCALE;
            s0.z = acc[m][t][2] * SCALE; s0.w = acc[m][t][3] * SCALE;
            s1.x = acc[m][t][4] * SCALE; s1.y = acc[m][t][5] * SCALE;
            s1.z = acc[m][t][6] * SCALE; s1.w = acc[m][t][7] * SCALE;
            __builtin_nontemporal_store(s0, (f32x4*)(az + (size_t)q * NN + kcol));
            __builtin_nontemporal_store(s1, (f32x4*)(az + (size_t)q * NN + kcol + 4));
        }
    }
}

// ---------------------------------------------------------------------------
// K5: in-place row softmax over attn[row][0..N).  wave32 shfl reductions +
// one 8-entry LDS combine (3 barriers total).
// ---------------------------------------------------------------------------
__global__ void k_softmax(float* __restrict__ attn) {
    const size_t row = blockIdx.x;
    float* p = attn + row * (size_t)NN;
    const int tid = threadIdx.x;             // 256 threads, 8 elems each
    const int wave = tid >> 5, lane = tid & 31;
    f32x4 v0 = *(f32x4*)(p + tid * 8);
    f32x4 v1 = *(f32x4*)(p + tid * 8 + 4);
    float m = fmaxf(fmaxf(fmaxf(v0.x, v0.y), fmaxf(v0.z, v0.w)),
                    fmaxf(fmaxf(v1.x, v1.y), fmaxf(v1.z, v1.w)));
    #pragma unroll
    for (int off = 16; off > 0; off >>= 1)
        m = fmaxf(m, __shfl_xor(m, off, 32));
    __shared__ float wred[8];
    if (lane == 0) wred[wave] = m;
    __syncthreads();
    m = wred[0];
    #pragma unroll
    for (int i = 1; i < 8; ++i) m = fmaxf(m, wred[i]);
    __syncthreads();                          // everyone read wred before reuse

    v0.x = __expf(v0.x - m); v0.y = __expf(v0.y - m);
    v0.z = __expf(v0.z - m); v0.w = __expf(v0.w - m);
    v1.x = __expf(v1.x - m); v1.y = __expf(v1.y - m);
    v1.z = __expf(v1.z - m); v1.w = __expf(v1.w - m);
    float s = v0.x + v0.y + v0.z + v0.w + v1.x + v1.y + v1.z + v1.w;
    #pragma unroll
    for (int off = 16; off > 0; off >>= 1)
        s += __shfl_xor(s, off, 32);
    if (lane == 0) wred[wave] = s;
    __syncthreads();
    s = wred[0];
    #pragma unroll
    for (int i = 1; i < 8; ++i) s += wred[i];

    const float inv = 1.0f / s;
    v0.x *= inv; v0.y *= inv; v0.z *= inv; v0.w *= inv;
    v1.x *= inv; v1.y *= inv; v1.z *= inv; v1.w *= inv;
    *(f32x4*)(p + tid * 8)     = v0;
    *(f32x4*)(p + tid * 8 + 4) = v1;
}

// ---------------------------------------------------------------------------
// K6: O = P @ V.  A = attn rows (fp32 -> bf16, streamed from HBM with
// global_prefetch two chunks ahead), B = V staged transposed into
// double-buffered LDS.  K-loop manually unrolled by 2 with the final two
// steps peeled, so the steady-state body is branch-free and every register
// index is a compile-time constant (no v_movrels, no loop versioning).
// Block: 256 thr = 8 waves, each wave one 16-row query tile; 128 queries/blk.
// ---------------------------------------------------------------------------
__global__ __launch_bounds__(256)
void k_pv(const float* __restrict__ attn, const __bf16* __restrict__ qkv,
          float* __restrict__ out) {
    const int z = blockIdx.y;
    const int b = z % BB, h = z / BB;
    const int q0 = blockIdx.x * 128;
    const int tid = threadIdx.x, wave = tid >> 5, lane = tid & 31;
    const float* az = attn + (size_t)z * NN * NN;
    const __bf16* vm = qkv + (size_t)b * NN * 3 * DIM
                           + (size_t)2 * NN * DIM + (size_t)h * HD;  // v rows, ld=DIM

    __shared__ __attribute__((aligned(16))) __bf16 vt[2][HD][40];  // V^T chunks
    f32x8 acc[4] = {};
    const int qrow = q0 + wave * 16;
    const float* arow = az + (size_t)qrow * NN;

    auto stage = [&](int buf, int k0) {   // V[k0..k0+31][0..63] -> vt[buf][n][k]
        for (int e = tid; e < 32 * (HD / 4); e += 256) {
            const int kk = e >> 4;          // 0..31
            const int nq = (e & 15) * 4;    // 0,4,...,60
            ushort4 v4 = *(const ushort4*)(vm + (size_t)(k0 + kk) * DIM + nq);
            vt[buf][nq + 0][kk] = __builtin_bit_cast(__bf16, v4.x);
            vt[buf][nq + 1][kk] = __builtin_bit_cast(__bf16, v4.y);
            vt[buf][nq + 2][kk] = __builtin_bit_cast(__bf16, v4.z);
            vt[buf][nq + 3][kk] = __builtin_bit_cast(__bf16, v4.w);
        }
    };
    auto compute = [&](int buf, bf16x16 a) {
        #pragma unroll
        for (int t = 0; t < 4; ++t) {
            bf16x16 bfrag = load_frag_bf16(&vt[buf][t * 16][0], 40, lane);
            acc[t] = wmma_bf16(a, bfrag, acc[t]);
        }
    };

    stage(0, 0);
    bf16x16 a0 = load_frag_f32(arow, NN, lane);
    bf16x16 a1;

    int k0 = 0;
    for (int it = 0; it < (NN / 32) - 2; it += 2) {   // steady state, branch-free
        __syncthreads();
        stage(1, k0 + 32);
        a1 = load_frag_f32(arow + k0 + 32, NN, lane);
        __builtin_prefetch(arow + k0 + 64 + (lane & 15) * NN, 0, 0);
        compute(0, a0);
        __syncthreads();
        stage(0, k0 + 64);
        a0 = load_frag_f32(arow + k0 + 64, NN, lane);
        __builtin_prefetch(arow + k0 + 96 + (lane & 15) * NN, 0, 0);
        compute(1, a1);
        k0 += 64;
    }
    // tail: steps (NSTEP-2, NSTEP-1), k0 == NN - 64
    __syncthreads();
    stage(1, k0 + 32);
    a1 = load_frag_f32(arow + k0 + 32, NN, lane);
    compute(0, a0);
    __syncthreads();
    compute(1, a1);

    const int half = lane >> 4, l15 = lane & 15;
    float* ob = out + (size_t)b * DIM * NN;
    #pragma unroll
    for (int t = 0; t < 4; ++t) {
        const int c = h * HD + t * 16 + l15;         // output channel
        f32x4 s0 = {acc[t][0], acc[t][1], acc[t][2], acc[t][3]};
        f32x4 s1 = {acc[t][4], acc[t][5], acc[t][6], acc[t][7]};
        __builtin_nontemporal_store(s0, (f32x4*)(ob + (size_t)c * NN + qrow + half * 8));
        __builtin_nontemporal_store(s1, (f32x4*)(ob + (size_t)c * NN + qrow + half * 8 + 4));
    }
}

// ---------------------------------------------------------------------------
extern "C" void kernel_launch(void* const* d_in, const int* in_sizes, int n_in,
                              void* d_out, int out_size, void* d_ws, size_t ws_size,
                              hipStream_t stream) {
    const float* x = (const float*)d_in[0];   // (B, DIM, N)
    const float* w = (const float*)d_in[1];   // (3*DIM, DIM)
    float* out  = (float*)d_out;                          // (B, DIM, N)
    float* attn = out + (size_t)BB * DIM * NN;            // (H*B, N, N)

    __bf16* xt  = (__bf16*)d_ws;                          // (B, N, DIM)
    __bf16* wb  = xt + (size_t)BB * NN * DIM;             // (3*DIM, DIM)
    __bf16* qkv = wb + (size_t)3 * DIM * DIM;             // (B, N, 3*DIM)

    k_transpose_x<<<dim3(NN / 32, DIM / 32, BB), dim3(32, 8), 0, stream>>>(x, xt);

    const int nw = 3 * DIM * DIM;
    k_convert_w<<<(nw + 255) / 256, 256, 0, stream>>>(w, wb, nw);

    k_qkv_gemm<<<dim3(3 * DIM / 64, NN / 256, BB), 256, 0, stream>>>(xt, wb, qkv);

    k_logits<<<dim3(NN / 64, NN / 256, HH * BB), 256, 0, stream>>>(qkv, attn);

    k_softmax<<<dim3(HH * BB * NN), 256, 0, stream>>>(attn);

    k_pv<<<dim3(NN / 128, HH * BB), 256, 0, stream>>>(attn, qkv, out);
}
